// HeteroRGCNLayer_58265526338114
// MI455X (gfx1250) — matrix operand order repaired and checked by reading, not compiled, not run on hardware
//
#include <hip/hip_runtime.h>

#define NA 8192
#define NB 8192
#define NE 32768
#define HW 256        // 16*16 spatial
#define FEAT 2048     // 8*256
#define KPAD 96       // 72 real k (9 taps * 8 cin) padded to 3 WMMA K-steps
#define BTS 104       // Bt row stride in halves (16B-aligned rows, bank-decorrelated)

typedef __attribute__((ext_vector_type(16))) __bf16         v16bf;
typedef __attribute__((ext_vector_type(8)))  float          v8f;
typedef __attribute__((ext_vector_type(16))) unsigned short v16us;
typedef __attribute__((ext_vector_type(8)))  unsigned short v8us;
typedef __attribute__((ext_vector_type(4)))  unsigned int   v4u;
typedef __attribute__((ext_vector_type(8)))  int            v8i;
typedef __attribute__((ext_vector_type(4)))  int            v4i;

typedef __attribute__((address_space(3))) void lds_void;

static __device__ __forceinline__ unsigned short f2bf(float f) {
  unsigned u = __builtin_bit_cast(unsigned, f);
  u += 0x7FFFu + ((u >> 16) & 1u);          // round-to-nearest-even
  return (unsigned short)(u >> 16);
}

// Load a 16-half WMMA fragment from two 16B LDS chunks.
static __device__ __forceinline__ v16bf ld_frag(const unsigned short* c0,
                                                const unsigned short* c1) {
  v8us lo = *(const v8us*)c0;
  v8us hi = *(const v8us*)c1;
  v16us v = __builtin_shufflevector(lo, hi, 0, 1, 2, 3, 4, 5, 6, 7,
                                            8, 9, 10, 11, 12, 13, 14, 15);
  return __builtin_bit_cast(v16bf, v);
}

// Issue one TDM copy: 8KB contiguous global -> LDS (1024 x 8B elements).
static __device__ __forceinline__ void tdm_copy_8k(unsigned lds_byte_off,
                                                   const float* gptr) {
  unsigned long long ga = (unsigned long long)(const void*)gptr;
  v4u g0;
  g0.x = 1u;                                          // count=1, user mode
  g0.y = lds_byte_off;                                // lds_addr
  g0.z = (unsigned)ga;                                // global_addr[31:0]
  g0.w = (unsigned)((ga >> 32) & 0x1FFFFFFu) | (2u << 30);  // addr[56:32] | type=2
  v8i g1;
  g1[0] = 0x30000;                // workgroup_mask=0, data_size=3 (8B)
  g1[1] = (int)(1024u << 16);     // tensor_dim0[15:0] = 1024
  g1[2] = (int)(1u << 16);        // tensor_dim0[31:16]=0, tensor_dim1[15:0]=1
  g1[3] = (int)(1024u << 16);     // tensor_dim1[31:16]=0, tile_dim0=1024
  g1[4] = 0;                      // tile_dim1=0 (unused), tile_dim2=0
  g1[5] = 1024;                   // tensor_dim0_stride[31:0]
  g1[6] = 0;
  g1[7] = 0;
  v4i z4 = {0, 0, 0, 0};
  v8i z8 = {0, 0, 0, 0, 0, 0, 0, 0};
  // clang-23 / therock-10.0 signature: (g0, g1, g2, g3, g4, cpol)
  __builtin_amdgcn_tensor_load_to_lds(g0, g1, z4, z4, z8, 0);
}

__global__ __launch_bounds__(256) void rgcn_edge_conv(
    const float* __restrict__ feat_a, const float* __restrict__ feat_b,
    const float* __restrict__ W1, const float* __restrict__ b1,
    const float* __restrict__ W2, const float* __restrict__ b2,
    const int* __restrict__ src1, const int* __restrict__ dst1,
    const int* __restrict__ src2, const int* __restrict__ dst2,
    float* __restrict__ out, float* __restrict__ cnt)
{
  __shared__ float uS[FEAT];                   // u  (TDM target; reused for m = u + Wh)
  __shared__ float vS[FEAT];                   // v  (TDM target)
  __shared__ unsigned short Bt[256 * BTS];     // bf16 im2col, row-major [pos][k]
  __shared__ unsigned short Aw[16 * KPAD];     // bf16 weights,  row-major [cout][k]

  const int blk = blockIdx.x;
  const bool et2 = blk >= NE;
  const int e = et2 ? blk - NE : blk;
  const int tid = threadIdx.x;

  const float* fS = et2 ? feat_b : feat_a;
  const float* fD = et2 ? feat_a : feat_b;
  const float* Wg = et2 ? W2 : W1;
  const float* bg = et2 ? b2 : b1;
  const int s  = (et2 ? src2 : src1)[e];
  const int dn = (et2 ? dst2 : dst1)[e];

  // d_out layout (floats): h_a | h_b | Wh1 | Wh2
  float* WhOut = out + (size_t)(NA + NB) * FEAT
                     + (et2 ? (size_t)NE * FEAT : (size_t)0)
                     + (size_t)e * FEAT;
  float* hAcc  = out + (et2 ? (size_t)0 : (size_t)NA * FEAT) + (size_t)dn * FEAT;
  float* cn    = cnt + (et2 ? 0 : NA) + dn;

  // ---- Stage 1: TDM async copies of u and v into LDS (one wave issues) ----
  if (__builtin_amdgcn_readfirstlane(tid) == 0) {   // wave-uniform guard (TDM ignores EXEC)
    unsigned lds_u = (unsigned)(size_t)(lds_void*)uS;
    unsigned lds_v = (unsigned)(size_t)(lds_void*)vS;
    tdm_copy_8k(lds_u, fS + (size_t)s  * FEAT);
    tdm_copy_8k(lds_v, fD + (size_t)dn * FEAT);
    __builtin_amdgcn_s_wait_tensorcnt(0);
  }

  // ---- Stage 2 (overlaps TDM): weight matrix Aw[16][96], k = tap*8 + cin ----
  for (int i = tid; i < 16 * KPAD; i += 256) {
    int m = i / KPAD, k = i - m * KPAD;
    int t = k >> 3, ci = k & 7;
    float val = (m < 8 && t < 9) ? Wg[(m * 8 + ci) * 9 + t] : 0.0f;
    Aw[i] = f2bf(val);
  }
  __syncthreads();

  // ---- Stage 3: bf16 im2col Bt[p][k] (ef = u + v on the fly), b128 stores ----
  {
    const int p = tid, h = p >> 4, w = p & 15;
#pragma unroll
    for (int t = 0; t < 12; ++t) {
      int kh = t / 3, kw = t - kh * 3;
      int hh = h + kh - 1, ww = w + kw - 1;
      bool ok = (t < 9) && ((unsigned)hh < 16u) && ((unsigned)ww < 16u);
      int sp = hh * 16 + ww;
      unsigned pk[4];
#pragma unroll
      for (int c2 = 0; c2 < 4; ++c2) {
        float v0 = ok ? (uS[(2 * c2)     * HW + sp] + vS[(2 * c2)     * HW + sp]) : 0.0f;
        float v1 = ok ? (uS[(2 * c2 + 1) * HW + sp] + vS[(2 * c2 + 1) * HW + sp]) : 0.0f;
        pk[c2] = (unsigned)f2bf(v0) | ((unsigned)f2bf(v1) << 16);
      }
      uint4 q = {pk[0], pk[1], pk[2], pk[3]};
      *(uint4*)&Bt[p * BTS + t * 8] = q;       // one ds_store_b128 per tap
    }
  }
  __syncthreads();

  // ---- Stage 4: WMMA implicit GEMM, 8 waves * 2 N-tiles * 3 K-steps ----
  const int wid  = tid >> 5, lane = tid & 31;
  const int row  = lane & 15;            // A row / B column within tile
  const int half = lane >> 4;            // half-wave selector

  // A fragments per ISA layout: lanes0-15 K{0..7,16..23}, lanes16-31 K{8..15,24..31}
  const int ao = half * 8;
  v16bf af[3];
#pragma unroll
  for (int ks = 0; ks < 3; ++ks)
    af[ks] = ld_frag(&Aw[row * KPAD + ks * 32 + ao],
                     &Aw[row * KPAD + ks * 32 + ao + 16]);

  float bias[8];
#pragma unroll
  for (int q = 0; q < 8; ++q) bias[q] = bg[q];

#pragma unroll
  for (int tt = 0; tt < 2; ++tt) {
    const int tile = wid + tt * 8;
    const int p = tile * 16 + row;
    const int bo = half * 8;             // lanes0-15 K{0..7,16..23}, lanes16-31 K{8..15,24..31}

    v8f acc = {0.f, 0.f, 0.f, 0.f, 0.f, 0.f, 0.f, 0.f};
#pragma unroll
    for (int ks = 0; ks < 3; ++ks) {
      v16bf bf = ld_frag(&Bt[p * BTS + ks * 32 + bo],
                         &Bt[p * BTS + ks * 32 + bo + 16]);
      acc = __builtin_amdgcn_wmma_f32_16x16x32_bf16(
          false, af[ks], false, bf, (short)0, acc, false, false);
    }

    // D layout: lane<16 holds column p, VGPR q = row cout=q (valid rows 0..7).
    if (lane < 16) {
#pragma unroll
      for (int q = 0; q < 8; ++q) {
        float wh = acc[q] + bias[q];
        WhOut[q * HW + p] = wh;                             // Wh output
        atomicAdd(&hAcc[q * HW + p], uS[q * HW + p] + wh);  // segment sum of m
      }
    }
  }

  if (tid == 0) atomicAdd(cn, 1.0f);
}

__global__ __launch_bounds__(256) void rgcn_finalize(float* __restrict__ out,
                                                     const float* __restrict__ cnt)
{
  size_t idx = (size_t)blockIdx.x * 256 + threadIdx.x;
  const size_t tot = (size_t)(NA + NB) * FEAT;
  if (idx >= tot) return;
  float c;
  if (idx < (size_t)NA * FEAT) c = cnt[idx >> 11];
  else                         c = cnt[NA + ((idx - (size_t)NA * FEAT) >> 11)];
  out[idx] = out[idx] / fmaxf(c, 1.0f);
}

extern "C" void kernel_launch(void* const* d_in, const int* in_sizes, int n_in,
                              void* d_out, int out_size, void* d_ws, size_t ws_size,
                              hipStream_t stream) {
  const float* feat_a = (const float*)d_in[0];
  const float* feat_b = (const float*)d_in[1];
  const float* W_e1   = (const float*)d_in[2];
  const float* b_e1   = (const float*)d_in[3];
  const float* W_e2   = (const float*)d_in[4];
  const float* b_e2   = (const float*)d_in[5];
  const int*   src_e1 = (const int*)d_in[6];
  const int*   dst_e1 = (const int*)d_in[7];
  const int*   src_e2 = (const int*)d_in[8];
  const int*   dst_e2 = (const int*)d_in[9];

  float* out = (float*)d_out;
  float* cnt = (float*)d_ws;   // [NA] counts for h_a, then [NB] counts for h_b

  // Zero the h accumulation regions and the counters (graph-capture safe).
  (void)hipMemsetAsync(out, 0, (size_t)(NA + NB) * FEAT * sizeof(float), stream);
  (void)hipMemsetAsync(cnt, 0, (size_t)(NA + NB) * sizeof(float), stream);

  rgcn_edge_conv<<<2 * NE, 256, 0, stream>>>(
      feat_a, feat_b, W_e1, b_e1, W_e2, b_e2,
      src_e1, dst_e1, src_e2, dst_e2, out, cnt);

  const size_t tot = (size_t)(NA + NB) * FEAT;
  rgcn_finalize<<<(unsigned)((tot + 255) / 256), 256, 0, stream>>>(out, cnt);
}
